// SelfAttention_56873956934338
// MI455X (gfx1250) — matrix operand order repaired and checked by reading, not compiled, not run on hardware
//
#include <hip/hip_runtime.h>

typedef __attribute__((ext_vector_type(16))) __bf16 v16bf;
typedef __attribute__((ext_vector_type(8)))  float  v8f;
typedef unsigned short u16;
typedef unsigned int   u32;
typedef unsigned long long u64;

#define N_BATCH 4
#define SEQ     2048
#define NHEADS  16
#define HD      64
#define EMB     1024
// per-tensor element count of projected q/k/v
#define QKV_STRIDE ((size_t)N_BATCH * NHEADS * SEQ * HD)

union FragU { v16bf v; u32 u[8]; u16 h[16]; };

__device__ __forceinline__ u16 f2bf(float f) {
  u32 u = __float_as_uint(f);
  u += 0x7FFFu + ((u >> 16) & 1u);   // round-to-nearest-even
  return (u16)(u >> 16);
}

// A-matrix (16x32 bf16) K-pair base for packed VGPR j, lane-half `half`
__device__ __forceinline__ int akp(int j, int half) {
  return (j < 4) ? (2 * j + 8 * half) : (16 + 2 * (j - 4) + 8 * half);
}

__device__ __forceinline__ v8f wmma_bf16(v16bf a, v16bf b, v8f c) {
  return __builtin_amdgcn_wmma_f32_16x16x32_bf16(
      false, a, false, b, (short)0, c, false, false);
}

// async 16-byte global -> LDS copy (per-lane), tracked by ASYNCcnt.
// LDS byte address = low 32 bits of the generic pointer (LDS aperture keeps
// the LDS offset in addr[31:0]).
__device__ __forceinline__ void async_ld16(const u16* gsrc, const u16* ldst) {
  u32 l = (u32)(uintptr_t)ldst;
  u64 g = (u64)(uintptr_t)gsrc;
  asm volatile("global_load_async_to_lds_b128 %0, %1, off"
               :: "v"(l), "v"(g) : "memory");
}
__device__ __forceinline__ void wait_async0() {
  asm volatile("s_wait_asynccnt 0" ::: "memory");
}

// ---------------------------------------------------------------------------
// Kernel 1: per-head QKV projection.  Q,K -> [t][N][H][S][D]; V -> transposed
// [N][H][D][S] so attention's P*V B-fragments are contiguous.
// grid = (1024, 3), block = 256.
// ---------------------------------------------------------------------------
__global__ __launch_bounds__(256) void qkv_project_kernel(
    const float* __restrict__ Xq, const float* __restrict__ Xk,
    const float* __restrict__ Xv, const float* __restrict__ Wq,
    const float* __restrict__ Wk, const float* __restrict__ Wv,
    u16* __restrict__ qkv) {
  __shared__ u16 Wlds[64 * 64];       // W[e][d], bf16
  __shared__ u16 Alds[8][16 * 64];    // per-wave 16 input rows, bf16

  const int t = blockIdx.y;
  const float* __restrict__ X = (t == 0) ? Xq : (t == 1) ? Xk : Xv;
  const float* __restrict__ W = (t == 0) ? Wq : (t == 1) ? Wk : Wv;

  const int tid = threadIdx.x;
  const int wave = tid >> 5, lane = tid & 31;
  const int half = lane >> 4, ln = lane & 15;

  for (int i = tid; i < 64 * 64; i += 256) Wlds[i] = f2bf(W[i]);

  const long row0 = ((long)blockIdx.x * 8 + wave) * 16;  // flat (n,s,h) row
  for (int i = lane; i < 16 * 64; i += 32)
    Alds[wave][i] = f2bf(X[(size_t)row0 * 64 + i]);
  __syncthreads();

  FragU a[2];
#pragma unroll
  for (int ks = 0; ks < 2; ks++)
#pragma unroll
    for (int j = 0; j < 8; j++)
      a[ks].u[j] = *(const u32*)&Alds[wave][ln * 64 + 16 * half + 2 * j + 32 * ks];

  v8f acc[4] = {v8f{}, v8f{}, v8f{}, v8f{}};
#pragma unroll
  for (int ks = 0; ks < 2; ks++) {
#pragma unroll
    for (int nt = 0; nt < 4; nt++) {
      FragU b;  // B(k=d, n=e) = W[e][d]; 16 consecutive d starting at 16*half
#pragma unroll
      for (int j = 0; j < 8; j++)
        b.u[j] = *(const u32*)&Wlds[(nt * 16 + ln) * 64 + 16 * half + 2 * j + 32 * ks];
      acc[nt] = wmma_bf16(a[ks].v, b.v, acc[nt]);
    }
  }

#pragma unroll
  for (int nt = 0; nt < 4; nt++)
#pragma unroll
    for (int r = 0; r < 8; r++) {
      long gr = row0 + r + 8 * half;        // flat row = (n*SEQ + s)*NHEADS + h
      int h = (int)(gr & 15);
      int s = (int)((gr >> 4) & (SEQ - 1));
      int n = (int)(gr >> 15);
      int d = nt * 16 + ln;
      size_t off;
      if (t == 2)  // V transposed: [N][H][D][S]
        off = 2 * QKV_STRIDE + (((size_t)n * NHEADS + h) * HD + d) * SEQ + s;
      else
        off = (((size_t)t * N_BATCH + n) * NHEADS + h) * ((size_t)SEQ * HD) +
              (size_t)s * HD + d;
      qkv[off] = f2bf(acc[nt][r]);
    }
}

// ---------------------------------------------------------------------------
// Kernel 2: flash attention.  One wave = one 16-query tile; all 8 waves of a
// block share (n,h) and consume K/V chunks staged once per block into LDS via
// double-buffered async global->LDS copies.
// grid = (SEQ/16/8 = 16, N*H = 64), block = 256.
// ---------------------------------------------------------------------------
__global__ __launch_bounds__(256) void attention_kernel(
    const u16* __restrict__ qkv, u16* __restrict__ attn_out) {
  __shared__ u16 Klds[2][32 * 64];   // double-buffered K chunk [key][d]
  __shared__ u16 Vlds[2][64 * 32];   // double-buffered V^T chunk [d][key]
  __shared__ u16 Plds[8][16 * 32];   // per-wave P tile

  const int tid = threadIdx.x;
  const int wave = tid >> 5, lane = tid & 31;
  const int half = lane >> 4, ln = lane & 15;

  const int nh = blockIdx.y;
  const int n = nh >> 4, h = nh & 15;
  const int qtile = blockIdx.x * 8 + wave;

  const size_t headoff = ((size_t)n * NHEADS + h) * SEQ * HD;
  const u16* __restrict__ Q  = qkv + headoff + (size_t)qtile * 16 * HD;
  const u16* __restrict__ K  = qkv + QKV_STRIDE + headoff;
  const u16* __restrict__ Vt = qkv + 2 * QKV_STRIDE + headoff;  // [d][s]

  // Q A-fragments (16 rows x K=64)
  FragU aq[2];
#pragma unroll
  for (int ks = 0; ks < 2; ks++)
#pragma unroll
    for (int j = 0; j < 8; j++)
      aq[ks].u[j] = *(const u32*)&Q[(size_t)ln * HD + 16 * half + 2 * j + 32 * ks];

  v8f o[4] = {v8f{}, v8f{}, v8f{}, v8f{}};
  float mi[8], li[8];
#pragma unroll
  for (int r = 0; r < 8; r++) { mi[r] = -1e30f; li[r] = 0.0f; }
  const float scale = 0.03125f;  // 1/sqrt(EMBED)

  const int vd = tid >> 2, vseg = tid & 3;  // V staging: 4 threads per d-row

  // prologue: stage chunk 0 into buffer 0 (each thread copies 16B of K, 16B of V)
  async_ld16(K + tid * 8, &Klds[0][tid * 8]);
  async_ld16(Vt + (size_t)vd * SEQ + vseg * 8, &Vlds[0][vd * 32 + vseg * 8]);
  wait_async0();
  __syncthreads();

  int buf = 0;
  for (int kc = 0; kc < SEQ / 32; kc++) {
    const int nbuf = buf ^ 1;
    if (kc + 1 < SEQ / 32) {  // prefetch next chunk while computing this one
      async_ld16(K + (size_t)(kc + 1) * 32 * HD + tid * 8, &Klds[nbuf][tid * 8]);
      async_ld16(Vt + (size_t)vd * SEQ + (kc + 1) * 32 + vseg * 8,
                 &Vlds[nbuf][vd * 32 + vseg * 8]);
    }

    // scores: two 16x16 tiles over K=64, B from LDS (ds_load_b128)
    v8f s0 = v8f{}, s1 = v8f{};
#pragma unroll
    for (int ks = 0; ks < 2; ks++) {
      FragU b0, b1;
#pragma unroll
      for (int j = 0; j < 8; j++) {
        b0.u[j] = *(const u32*)&Klds[buf][ln * 64 + 16 * half + 2 * j + 32 * ks];
        b1.u[j] = *(const u32*)&Klds[buf][(16 + ln) * 64 + 16 * half + 2 * j + 32 * ks];
      }
      s0 = wmma_bf16(aq[ks].v, b0.v, s0);
      s1 = wmma_bf16(aq[ks].v, b1.v, s1);
    }

    // online softmax (row stats replicated across each 16-lane half)
    float mloc[8];
#pragma unroll
    for (int r = 0; r < 8; r++) {
      s0[r] *= scale; s1[r] *= scale;
      mloc[r] = fmaxf(s0[r], s1[r]);
    }
#pragma unroll
    for (int off = 1; off < 16; off <<= 1)
#pragma unroll
      for (int r = 0; r < 8; r++)
        mloc[r] = fmaxf(mloc[r], __shfl_xor(mloc[r], off, 32));

    float p0[8], p1[8], rs[8], alpha[8];
#pragma unroll
    for (int r = 0; r < 8; r++) {
      float mn = fmaxf(mi[r], mloc[r]);
      alpha[r] = __expf(mi[r] - mn);
      p0[r] = __expf(s0[r] - mn);
      p1[r] = __expf(s1[r] - mn);
      rs[r] = p0[r] + p1[r];
      mi[r] = mn;
      li[r] *= alpha[r];
    }
#pragma unroll
    for (int off = 1; off < 16; off <<= 1)
#pragma unroll
      for (int r = 0; r < 8; r++)
        rs[r] += __shfl_xor(rs[r], off, 32);
#pragma unroll
    for (int r = 0; r < 8; r++) li[r] += rs[r];

#pragma unroll
    for (int nt = 0; nt < 4; nt++)
#pragma unroll
      for (int r = 0; r < 8; r++) o[nt][r] *= alpha[r];

    // repack P (C-layout) -> A-fragment layout via per-wave LDS slab
    u16* P = Plds[wave];
#pragma unroll
    for (int r = 0; r < 8; r++) {
      int row = r + 8 * half;
      P[row * 32 + ln]      = f2bf(p0[r]);
      P[row * 32 + 16 + ln] = f2bf(p1[r]);
    }
    FragU ap;  // same-wave LDS ops are in-order
#pragma unroll
    for (int j = 0; j < 8; j++)
      ap.u[j] = *(const u32*)&P[ln * 32 + 16 * half + 2 * j];

    // P(16x32) * V(32x64): B from transposed-V LDS, contiguous key pairs
#pragma unroll
    for (int nt = 0; nt < 4; nt++) {
      FragU bv;
#pragma unroll
      for (int j = 0; j < 8; j++)
        bv.u[j] = *(const u32*)&Vlds[buf][(nt * 16 + ln) * 32 + 16 * half + 2 * j];
      o[nt] = wmma_bf16(ap.v, bv.v, o[nt]);
    }

    wait_async0();     // next buffers resident in LDS
    __syncthreads();   // everyone done reading current buffers
    buf = nbuf;
  }

  // normalize and write attn output [N*S][EMB] bf16 (col = h*64 + d)
#pragma unroll
  for (int nt = 0; nt < 4; nt++)
#pragma unroll
    for (int r = 0; r < 8; r++) {
      size_t rowg = (size_t)n * SEQ + qtile * 16 + r + 8 * half;
      int col = h * HD + nt * 16 + ln;
      attn_out[rowg * EMB + col] = f2bf(o[nt][r] / li[r]);
    }
}

// ---------------------------------------------------------------------------
// Kernel A: one-time Wo fp32 -> bf16 conversion (shared by all blocks of k3).
// ---------------------------------------------------------------------------
__global__ __launch_bounds__(256) void convert_wo_kernel(
    const float* __restrict__ Wo, u16* __restrict__ wo_bf) {
  size_t i = ((size_t)blockIdx.x * 256 + threadIdx.x) * 4;
#pragma unroll
  for (int j = 0; j < 4; j++) wo_bf[i + j] = f2bf(Wo[i + j]);
}

// ---------------------------------------------------------------------------
// Kernel 3: output projection  out = attn @ Wo^T + bo  (fp32 out).
// Wo tiles (64 x 32, bf16) async-staged into LDS, double-buffered, shared by
// the 8 waves of the block.  grid = (64, 16), block = 256.
// ---------------------------------------------------------------------------
__global__ __launch_bounds__(256) void out_proj_kernel(
    const u16* __restrict__ attn, const u16* __restrict__ wo_bf,
    const float* __restrict__ bo, float* __restrict__ out) {
  __shared__ u16 Wbuf[2][64 * 32];

  const int tid = threadIdx.x;
  const int wave = tid >> 5, lane = tid & 31;
  const int half = lane >> 4, ln = lane & 15;

  const long row0 = ((long)blockIdx.x * 8 + wave) * 16;
  const int col0 = blockIdx.y * 64;

  const int we = tid >> 2, wseg = tid & 3;  // 4 threads per weight row

  async_ld16(wo_bf + (size_t)(col0 + we) * EMB + wseg * 8,
             &Wbuf[0][we * 32 + wseg * 8]);
  wait_async0();
  __syncthreads();

  v8f acc[4] = {v8f{}, v8f{}, v8f{}, v8f{}};
  int buf = 0;
  for (int ks = 0; ks < EMB / 32; ks++) {
    const int nbuf = buf ^ 1;
    if (ks + 1 < EMB / 32)
      async_ld16(wo_bf + (size_t)(col0 + we) * EMB + (ks + 1) * 32 + wseg * 8,
                 &Wbuf[nbuf][we * 32 + wseg * 8]);

    FragU a;
#pragma unroll
    for (int j = 0; j < 8; j++)
      a.u[j] = *(const u32*)&attn[(size_t)(row0 + ln) * EMB + 16 * half + 2 * j + 32 * ks];
#pragma unroll
    for (int nt = 0; nt < 4; nt++) {
      FragU b;
#pragma unroll
      for (int j = 0; j < 8; j++)
        b.u[j] = *(const u32*)&Wbuf[buf][(nt * 16 + ln) * 32 + 16 * half + 2 * j];
      acc[nt] = wmma_bf16(a.v, b.v, acc[nt]);
    }

    wait_async0();
    __syncthreads();
    buf = nbuf;
  }

#pragma unroll
  for (int nt = 0; nt < 4; nt++) {
    int col = col0 + nt * 16 + ln;
    float bias = bo[col];
#pragma unroll
    for (int r = 0; r < 8; r++)
      out[(size_t)(row0 + r + 8 * half) * EMB + col] = acc[nt][r] + bias;
  }
}

extern "C" void kernel_launch(void* const* d_in, const int* in_sizes, int n_in,
                              void* d_out, int out_size, void* d_ws,
                              size_t ws_size, hipStream_t stream) {
  const float* Xq = (const float*)d_in[0];
  const float* Xk = (const float*)d_in[1];
  const float* Xv = (const float*)d_in[2];
  const float* Wq = (const float*)d_in[3];
  const float* Wk = (const float*)d_in[4];
  const float* Wv = (const float*)d_in[5];
  const float* Wo = (const float*)d_in[6];
  const float* bo = (const float*)d_in[7];

  u16* qkv   = (u16*)d_ws;                                           // 48 MB
  u16* attn  = (u16*)((char*)d_ws + 3 * QKV_STRIDE * sizeof(u16));   // 16 MB
  u16* wo_bf = attn + (size_t)N_BATCH * SEQ * EMB;                   //  2 MB

  dim3 blk(256);
  qkv_project_kernel<<<dim3(1024, 3), blk, 0, stream>>>(Xq, Xk, Xv, Wq, Wk, Wv, qkv);
  convert_wo_kernel<<<dim3(1024), blk, 0, stream>>>(Wo, wo_bf);
  attention_kernel<<<dim3(16, 64), blk, 0, stream>>>(qkv, attn);
  out_proj_kernel<<<dim3(64, 16), blk, 0, stream>>>(attn, wo_bf, bo, (float*)d_out);
}
